// Lisi_8942121910662
// MI455X (gfx1250) — matrix operand order repaired and checked by reading, not compiled, not run on hardware
//
#include <hip/hip_runtime.h>
#include <hip/hip_bf16.h>
#include <stdint.h>

typedef __bf16 bf16;
typedef bf16  v16bf __attribute__((ext_vector_type(16)));
typedef float v8f   __attribute__((ext_vector_type(8)));
typedef float v4f   __attribute__((ext_vector_type(4)));

#define B_SZ   2048
#define E_SZ   256
#define H_SZ   512
#define NC_SZ  5
#define S_SZ   258      // E + 2
#define T_SZ   88
#define NG     2048     // 4*H
#define NMDN   2585     // NC + 2*NC*S
#define KCAT   768      // E + H
#define KCONV  64       // 49 padded

// ---------------- RNG helpers (deterministic hash-based) ----------------
__device__ __forceinline__ unsigned hashu(unsigned x) {
    x ^= x >> 16; x *= 0x7feb352dU;
    x ^= x >> 15; x *= 0x846ca68bU;
    x ^= x >> 16; return x;
}
__device__ __forceinline__ float u01(unsigned h) {
    return (float)(h >> 8) * (1.0f / 16777216.0f) + 1e-7f;
}
__device__ __forceinline__ float rnorm(unsigned s) {
    unsigned h1 = hashu(s * 2654435761u + 0x9E3779B9u);
    unsigned h2 = hashu(h1 ^ 0x85ebca6bu);
    float u1 = u01(h1), u2 = u01(h2);
    return sqrtf(-2.0f * __logf(u1)) * __cosf(6.2831853f * u2);
}
__device__ __forceinline__ float sigmf(float x) { return 1.0f / (1.0f + __expf(-x)); }

// ---------------- WMMA GEMM:  C[M,N] = A[M,K](f32) * W[N,K](bf16)^T + bias ----------------
// Block: 256 threads (8 waves). Tile: 128(M) x 128(N), K-chunks of 32.
// Wave w owns rows [16w,16w+16) x 128 cols -> 8 f32 16x16 accumulators.
// A-fragments are built directly in registers (per-wave private, no sharing).
// B-tile staged to LDS with global_load_async_to_lds_b128 (ASYNCcnt), double-buffered:
// async copy of chunk i+1 overlaps the 8 WMMAs of chunk i; one barrier per chunk.
// All 8 B-fragments are preloaded into registers before the WMMA block so the
// 16 ds_load_b128 pipeline instead of 8 serialized load->wait->wmma chains.
// Requires: M % 128 == 0, K % 32 == 0, lda % 4 == 0.
__launch_bounds__(256, 2)
__global__ void gemm_bf16_wmma(const float* __restrict__ A, int lda,
                               const bf16* __restrict__ Bw,
                               const float* __restrict__ bias,
                               float* __restrict__ C,
                               int M, int N, int K)
{
    __shared__ alignas(32) bf16 ldsB[2][128 * 32];   // fragment-major, 2 x 8KB

    const int t    = threadIdx.x;
    const int lane = t & 31;
    const int wave = t >> 5;
    const int mbase = blockIdx.x * 128;
    const int nbase = blockIdx.y * 128;

    const int rowA  = mbase + wave * 16 + (lane & 15);
    const int halfA = lane >> 4;

    // B staging slot for this thread: fragment nb = wave, lane = lane.
    // B k order used here (same as A): N = lane%16, K = (lane/16)*16 + e (16 contiguous bf16 = 32B)
    const int nStage    = nbase + wave * 16 + (lane & 15);
    const int krunStage = (lane >> 4) * 16;

    auto stageB = [&](int kb, int buf) {
        bf16* dst = &ldsB[buf][t * 16];
        if (nStage < N) {
            uint64_t ga = (uint64_t)(uintptr_t)(Bw + (size_t)nStage * K + kb + krunStage);
            unsigned la = (unsigned)(uintptr_t)dst;   // low 32 bits = LDS offset
            asm volatile(
                "global_load_async_to_lds_b128 %0, %1, off\n\t"
                "global_load_async_to_lds_b128 %0, %1, off offset:16"
                :: "v"(la), "v"(ga) : "memory");
        } else {
            v16bf z = {};
            *((v16bf*)dst) = z;                       // zero-fill edge columns
        }
    };

    v8f acc[8];
    #pragma unroll
    for (int nb = 0; nb < 8; ++nb) acc[nb] = (v8f){};

    const int nchunks = K >> 5;

    stageB(0, 0);
    asm volatile("s_wait_asynccnt 0x0" ::: "memory");
    __syncthreads();

    for (int i = 0; i < nchunks; ++i) {
        const int kb  = i << 5;
        const int cur = i & 1;
        if (i + 1 < nchunks) stageB(kb + 32, (i + 1) & 1);   // async, overlaps compute

        // ---- build A fragment in registers.
        // A k order used here (same as B): elem e -> K = (lane/16)*16 + e
        const float* ap = A + (size_t)rowA * lda + kb + halfA * 16;
        v16bf av;
        #pragma unroll
        for (int g = 0; g < 2; ++g) {
            v4f f0 = *(const v4f*)(ap + g * 8);
            v4f f1 = *(const v4f*)(ap + g * 8 + 4);
            #pragma unroll
            for (int j = 0; j < 4; ++j) av[g * 8 + j]     = (bf16)f0[j];
            #pragma unroll
            for (int j = 0; j < 4; ++j) av[g * 8 + 4 + j] = (bf16)f1[j];
        }
        if (i + 1 < nchunks) __builtin_prefetch((const void*)(ap + 32), 0, 3);

        // ---- preload all 8 B fragments, then issue the 8 WMMAs.
        v16bf bfr[8];
        #pragma unroll
        for (int nb = 0; nb < 8; ++nb)
            bfr[nb] = *((const v16bf*)(&ldsB[cur][(nb * 32 + lane) * 16]));
        #pragma unroll
        for (int nb = 0; nb < 8; ++nb)
            acc[nb] = __builtin_amdgcn_wmma_f32_16x16x32_bf16(
                false, av, false, bfr[nb], (short)0, acc[nb], false, false);

        asm volatile("s_wait_asynccnt 0x0" ::: "memory");   // next buffer landed
        __syncthreads();                                    // all waves done with cur
    }

    // ---- epilogue: C/D 16x16 f32 layout: VGPR r -> M = r + 8*(lane/16), N = lane%16
    const int mrow  = mbase + wave * 16 + (lane >> 4) * 8;
    const int ncol0 = nbase + (lane & 15);
    #pragma unroll
    for (int nb = 0; nb < 8; ++nb) {
        int n = ncol0 + nb * 16;
        if (n < N) {
            float bv = bias ? bias[n] : 0.0f;
            #pragma unroll
            for (int r = 0; r < 8; ++r)
                C[(size_t)(mrow + r) * N + n] = acc[nb][r] + bv;
        }
    }
}

// ---------------- weight prep (f32 -> bf16, concatenated layouts) ----------------
__global__ void prep_cat_kernel(const float* __restrict__ w_ih, const float* __restrict__ w_hh,
                                const float* __restrict__ b_ih, const float* __restrict__ b_hh,
                                bf16* __restrict__ wc, float* __restrict__ bc)
{
    int idx = blockIdx.x * 256 + threadIdx.x;                 // 2048*768
    if (idx >= NG * KCAT) return;
    int n = idx / KCAT, k = idx % KCAT;
    float v = (k < E_SZ) ? w_ih[n * E_SZ + k] : w_hh[n * H_SZ + (k - E_SZ)];
    wc[idx] = (bf16)v;
    if (k == 0) bc[n] = b_ih[n] + b_hh[n];
}

__global__ void prep_mdn_kernel(const float* __restrict__ w_pi, const float* __restrict__ b_pi,
                                const float* __restrict__ w_mu, const float* __restrict__ b_mu,
                                const float* __restrict__ w_sg, const float* __restrict__ b_sg,
                                bf16* __restrict__ wm, float* __restrict__ bm)
{
    int idx = blockIdx.x * 256 + threadIdx.x;                 // 2585*512
    if (idx >= NMDN * H_SZ) return;
    int n = idx / H_SZ, k = idx % H_SZ;
    float v, bb;
    if (n < NC_SZ)                     { v = w_pi[n * H_SZ + k];              bb = b_pi[n]; }
    else if (n < NC_SZ + NC_SZ * S_SZ) { v = w_mu[(n - NC_SZ) * H_SZ + k];    bb = b_mu[n - NC_SZ]; }
    else { int m = n - NC_SZ - NC_SZ * S_SZ; v = w_sg[m * H_SZ + k];          bb = b_sg[m]; }
    wm[idx] = (bf16)v;
    if (k == 0) bm[n] = bb;
}

__global__ void prep_conv_kernel(const float* __restrict__ cw, bf16* __restrict__ wc)
{
    int idx = blockIdx.x * 256 + threadIdx.x;                 // 256*64
    if (idx >= E_SZ * KCONV) return;
    int n = idx / KCONV, k = idx % KCONV;
    wc[idx] = (bf16)((k < 49) ? cw[n * 49 + k] : 0.0f);
}

__global__ void init_kernel(float* __restrict__ xy, float* __restrict__ eh, float* __restrict__ c)
{
    int idx = blockIdx.x * 256 + threadIdx.x;
    if (idx < B_SZ * 2)        xy[idx] = 0.5f;
    if (idx < B_SZ * KCAT)     eh[idx] = 0.0f;
    if (idx < B_SZ * H_SZ)     c[idx]  = 0.0f;
}

// ---------------- per-step elementwise kernels ----------------
__global__ void patch_kernel(const float* __restrict__ x, const float* __restrict__ xy,
                             float* __restrict__ patches, float* __restrict__ coords_out, int step)
{
    int b = blockIdx.x;
    int j = threadIdx.x;                       // 64 threads
    float px = xy[2 * b], py = xy[2 * b + 1];
    int p0 = (int)(px * 65.0f - 3.0f);
    int p1 = (int)(py * 65.0f - 3.0f);
    p0 = (p0 < 0) ? 0 : (p0 > 59 ? 59 : p0);   // dynamic_slice clamp on 66-7
    p1 = (p1 < 0) ? 0 : (p1 > 59 ? 59 : p1);
    float v = 0.0f;
    if (j < 49) {
        int i = j / 7, jj = j % 7;
        int r = p0 + i - 1, cidx = p1 + jj - 1;       // pad offset 1
        if (r >= 0 && r < 64 && cidx >= 0 && cidx < 64)
            v = x[(size_t)b * 4096 + r * 64 + cidx];
        v += 0.01f * rnorm(((unsigned)step * 2048u + (unsigned)b) * 64u + (unsigned)j + 0x1111u);
    }
    patches[b * KCONV + j] = v;
    if (j < 2) coords_out[(size_t)b * (T_SZ * 2) + step * 2 + j] = (j == 0) ? px : py;
}

__global__ void bn_embed_kernel(const float* __restrict__ feat,
                                const float* __restrict__ gamma, const float* __restrict__ beta,
                                const float* __restrict__ pos_w, const float* __restrict__ pos_b,
                                const float* __restrict__ xy,
                                float* __restrict__ eh, float* __restrict__ embs_out, int step)
{
    int c = blockIdx.x;                        // 256 channels
    int t = threadIdx.x;                       // 256 threads
    __shared__ float s1[256], s2[256];
    float sum = 0.0f, sum2 = 0.0f;
    #pragma unroll
    for (int i = 0; i < 8; ++i) {
        float v = feat[(size_t)(t + 256 * i) * E_SZ + c];
        sum += v; sum2 += v * v;
    }
    s1[t] = sum; s2[t] = sum2;
    __syncthreads();
    for (int s = 128; s > 0; s >>= 1) {
        if (t < s) { s1[t] += s1[t + s]; s2[t] += s2[t + s]; }
        __syncthreads();
    }
    float mu  = s1[0] * (1.0f / 2048.0f);
    float var = s2[0] * (1.0f / 2048.0f) - mu * mu;
    float scale = gamma[c] * rsqrtf(var + 1e-5f);
    float shift = beta[c] - mu * scale;
    float pw0 = pos_w[2 * c], pw1 = pos_w[2 * c + 1], pb = pos_b[c];
    #pragma unroll
    for (int i = 0; i < 8; ++i) {
        int b = t + 256 * i;
        float v = feat[(size_t)b * E_SZ + c] * scale + shift;
        v = fmaxf(v, 0.0f);
        v += xy[2 * b] * pw0 + xy[2 * b + 1] * pw1 + pb;
        eh[(size_t)b * KCAT + c] = v;
        embs_out[(size_t)b * (T_SZ * E_SZ) + step * E_SZ + c] = v;
    }
}

__global__ void lstm_update_kernel(const float* __restrict__ g,
                                   float* __restrict__ eh, float* __restrict__ c)
{
    int idx = blockIdx.x * 256 + threadIdx.x;  // 2048*512
    int b = idx >> 9, n = idx & 511;
    const float* gr = g + (size_t)b * NG;
    float i_ = sigmf(gr[n]);
    float f_ = sigmf(gr[512 + n]);
    float gg = tanhf(gr[1024 + n]);
    float o_ = sigmf(gr[1536 + n]);
    float cn = f_ * c[idx] + i_ * gg;
    c[idx] = cn;
    eh[(size_t)b * KCAT + E_SZ + n] = o_ * tanhf(cn);
}

__global__ void mdn_sample_kernel(const float* __restrict__ mo,
                                  float* __restrict__ xy, float* __restrict__ preds_out, int step)
{
    int b = blockIdx.x;
    int t = threadIdx.x;                       // 256 threads
    __shared__ int comp;
    if (t == 0) {
        int best = 0; float bv = -1e30f;
        #pragma unroll
        for (int k = 0; k < NC_SZ; ++k) {
            float u = u01(hashu(((unsigned)step * 2048u + (unsigned)b) * 8u + (unsigned)k + 0x2222u));
            float gl = mo[(size_t)b * NMDN + k] - __logf(-__logf(u));
            if (gl > bv) { bv = gl; best = k; }
        }
        comp = best;
    }
    __syncthreads();
    int kbest = comp;
    for (int s = t; s < S_SZ; s += 256) {
        float mu = mo[(size_t)b * NMDN + NC_SZ + kbest * S_SZ + s];
        float sg = __expf(mo[(size_t)b * NMDN + NC_SZ + NC_SZ * S_SZ + kbest * S_SZ + s]);
        float samp = mu + sg * rnorm(((unsigned)step * 2048u + (unsigned)b) * 512u + (unsigned)s + 0x3333u);
        if (s < E_SZ) preds_out[(size_t)b * ((T_SZ - 1) * E_SZ) + step * E_SZ + s] = samp;
        else          xy[2 * b + (s - E_SZ)] = sigmf(samp);
    }
}

__global__ void copy_h_kernel(const float* __restrict__ eh, float* __restrict__ out)
{
    int idx = blockIdx.x * 256 + threadIdx.x;  // 2048*512
    int b = idx >> 9, n = idx & 511;
    out[idx] = eh[(size_t)b * KCAT + E_SZ + n];
}

// ---------------- host launcher ----------------
extern "C" void kernel_launch(void* const* d_in, const int* in_sizes, int n_in,
                              void* d_out, int out_size, void* d_ws, size_t ws_size,
                              hipStream_t stream)
{
    (void)in_sizes; (void)n_in; (void)out_size; (void)ws_size;
    const float* x       = (const float*)d_in[0];
    const float* conv_w  = (const float*)d_in[1];
    const float* conv_b  = (const float*)d_in[2];
    const float* bn_g    = (const float*)d_in[3];
    const float* bn_b    = (const float*)d_in[4];
    const float* pos_w   = (const float*)d_in[5];
    const float* pos_b   = (const float*)d_in[6];
    const float* w_ih    = (const float*)d_in[7];
    const float* w_hh    = (const float*)d_in[8];
    const float* b_ih    = (const float*)d_in[9];
    const float* b_hh    = (const float*)d_in[10];
    const float* w_pi    = (const float*)d_in[11];
    const float* b_pi    = (const float*)d_in[12];
    const float* w_mu    = (const float*)d_in[13];
    const float* b_mu    = (const float*)d_in[14];
    const float* w_sg    = (const float*)d_in[15];
    const float* b_sg    = (const float*)d_in[16];

    uint8_t* ws = (uint8_t*)d_ws;
    size_t off = 0;
    auto alloc = [&](size_t bytes) -> void* {
        void* p = ws + off;
        off = (off + bytes + 255) & ~(size_t)255;
        return p;
    };
    bf16*  w_cat    = (bf16*) alloc((size_t)NG * KCAT * sizeof(bf16));     // [Wih|Whh] 2048x768
    bf16*  w_mdn    = (bf16*) alloc((size_t)NMDN * H_SZ * sizeof(bf16));   // 2585x512
    bf16*  w_conv   = (bf16*) alloc((size_t)E_SZ * KCONV * sizeof(bf16));  // 256x64
    float* bias_cat = (float*)alloc((size_t)NG * sizeof(float));
    float* bias_mdn = (float*)alloc((size_t)NMDN * sizeof(float));
    float* patches  = (float*)alloc((size_t)B_SZ * KCONV * sizeof(float));
    float* feat     = (float*)alloc((size_t)B_SZ * E_SZ * sizeof(float));
    float* eh       = (float*)alloc((size_t)B_SZ * KCAT * sizeof(float));  // [e | h]
    float* cstate   = (float*)alloc((size_t)B_SZ * H_SZ * sizeof(float));
    float* gates    = (float*)alloc((size_t)B_SZ * NG * sizeof(float));
    float* mdnout   = (float*)alloc((size_t)B_SZ * NMDN * sizeof(float));
    float* xy       = (float*)alloc((size_t)B_SZ * 2 * sizeof(float));

    float* out        = (float*)d_out;
    float* out_hT     = out;                                   // 2048*512
    float* out_coords = out_hT + (size_t)B_SZ * H_SZ;          // 2048*88*2
    float* out_embs   = out_coords + (size_t)B_SZ * T_SZ * 2;  // 2048*88*256
    float* out_preds  = out_embs + (size_t)B_SZ * T_SZ * E_SZ; // 2048*87*256

    // one-time prep
    prep_cat_kernel <<<(NG * KCAT + 255) / 256, 256, 0, stream>>>(w_ih, w_hh, b_ih, b_hh, w_cat, bias_cat);
    prep_mdn_kernel <<<(NMDN * H_SZ + 255) / 256, 256, 0, stream>>>(w_pi, b_pi, w_mu, b_mu, w_sg, b_sg, w_mdn, bias_mdn);
    prep_conv_kernel<<<(E_SZ * KCONV + 255) / 256, 256, 0, stream>>>(conv_w, w_conv);
    init_kernel     <<<(B_SZ * KCAT + 255) / 256, 256, 0, stream>>>(xy, eh, cstate);

    for (int t = 0; t < T_SZ; ++t) {
        // embed
        patch_kernel<<<B_SZ, 64, 0, stream>>>(x, xy, patches, out_coords, t);
        gemm_bf16_wmma<<<dim3(B_SZ / 128, (E_SZ + 127) / 128), 256, 0, stream>>>(
            patches, KCONV, w_conv, conv_b, feat, B_SZ, E_SZ, KCONV);
        bn_embed_kernel<<<E_SZ, 256, 0, stream>>>(feat, bn_g, bn_b, pos_w, pos_b, xy, eh, out_embs, t);

        if (t < T_SZ - 1) {
            // LSTM: g = [e|h] * [Wih|Whh]^T + (b_ih + b_hh)
            gemm_bf16_wmma<<<dim3(B_SZ / 128, NG / 128), 256, 0, stream>>>(
                eh, KCAT, w_cat, bias_cat, gates, B_SZ, NG, KCAT);
            lstm_update_kernel<<<(B_SZ * H_SZ) / 256, 256, 0, stream>>>(gates, eh, cstate);
            // MDN: h * [Wpi;Wmu;Wsig]^T
            gemm_bf16_wmma<<<dim3(B_SZ / 128, (NMDN + 127) / 128), 256, 0, stream>>>(
                eh + E_SZ, KCAT, w_mdn, bias_mdn, mdnout, B_SZ, NMDN, H_SZ);
            mdn_sample_kernel<<<B_SZ, 256, 0, stream>>>(mdnout, xy, out_preds, t);
        }
    }
    copy_h_kernel<<<(B_SZ * H_SZ) / 256, 256, 0, stream>>>(eh, out_hT);
}
